// CustomTransformerEncoderLayer_83227876262150
// MI455X (gfx1250) — compile-verified, matrix operand order checked by reading, not run on hardware
//
#include <hip/hip_runtime.h>
#include <hip/hip_bf16.h>
#include <cstdint>

// ---------------------------------------------------------------------------
// Types for CDNA5 WMMA (wave32, 16x16x32 bf16 -> f32)
// ---------------------------------------------------------------------------
typedef __attribute__((ext_vector_type(16))) __bf16 v16bf;
typedef __attribute__((ext_vector_type(8)))  __bf16 v8bf;
typedef __attribute__((ext_vector_type(8)))  float  v8f;
typedef __attribute__((address_space(3)))    __bf16 lds_bf16;

#define EMBED 768
#define HEADS 12
#define MLPD  3072
#define SEQ   256
#define BATCH 2
#define TOK   512      // SEQ*BATCH
#define QKVD  2304     // 3*EMBED
#define LDQKV 4608     // BATCH*QKVD (row stride over l for fixed n)

__device__ __forceinline__ v8f vzero8() {
  v8f z;
#pragma unroll
  for (int i = 0; i < 8; ++i) z[i] = 0.0f;
  return z;
}

__device__ __forceinline__ v8f wmma_bf16(v16bf a, v16bf b, v8f c) {
  // v_wmma_f32_16x16x32_bf16
  return __builtin_amdgcn_wmma_f32_16x16x32_bf16(false, a, false, b, (short)0, c,
                                                 false, false);
}

__device__ __forceinline__ void wait_asynccnt0() {
#if __has_builtin(__builtin_amdgcn_s_wait_asynccnt)
  __builtin_amdgcn_s_wait_asynccnt(0);
#else
  asm volatile("s_wait_asynccnt 0" ::: "memory");
#endif
}

// Per-lane async copy of 16B global -> LDS (ASYNCcnt-tracked).
__device__ __forceinline__ void async_g2l_b128(lds_bf16* lp, const __bf16* gp) {
  asm volatile("global_load_async_to_lds_b128 %0, %1, off"
               :: "v"(lp), "v"(gp)
               : "memory");
}

// A or B^T fragment from a row-major f32 matrix: element i holds
// K = (i/8)*16 + (lane&16 ? 8 : 0) + i%8 of row `row`. K-runs of 8 are
// contiguous -> two float4 loads per half.
__device__ __forceinline__ v16bf frag_row_f32(const float* base, long row, long ld,
                                              int k0, int lane) {
  const int khalf = (lane & 16) ? 8 : 0;
  const float* p = base + row * ld + k0 + khalf;
  v16bf r;
#pragma unroll
  for (int g = 0; g < 2; ++g) {
    float4 f0 = *(const float4*)(p + g * 16);
    float4 f1 = *(const float4*)(p + g * 16 + 4);
    r[8 * g + 0] = (__bf16)f0.x; r[8 * g + 1] = (__bf16)f0.y;
    r[8 * g + 2] = (__bf16)f0.z; r[8 * g + 3] = (__bf16)f0.w;
    r[8 * g + 4] = (__bf16)f1.x; r[8 * g + 5] = (__bf16)f1.y;
    r[8 * g + 6] = (__bf16)f1.z; r[8 * g + 7] = (__bf16)f1.w;
  }
  return r;
}

// Same fragment from a row-major bf16 matrix (global: b128; LDS: ds_load_b128).
__device__ __forceinline__ v16bf frag_row_bf16(const __bf16* base, long row, long ld,
                                               int k0, int lane) {
  const int khalf = (lane & 16) ? 8 : 0;
  const __bf16* p = base + row * ld + k0 + khalf;
  v16bf r;
#pragma unroll
  for (int g = 0; g < 2; ++g) {
    v8bf f = *(const v8bf*)(p + g * 16);
#pragma unroll
    for (int j = 0; j < 8; ++j) r[8 * g + j] = f[j];
  }
  return r;
}

// B fragment for C = A*B with B row-major [K,N] (non-transposed): gather.
__device__ __forceinline__ v16bf frag_nn_f32(const float* base, int k0, long ld,
                                             int col, int lane) {
  const int khalf = (lane & 16) ? 8 : 0;
  v16bf r;
#pragma unroll
  for (int g = 0; g < 2; ++g)
#pragma unroll
    for (int j = 0; j < 8; ++j)
      r[8 * g + j] = (__bf16)base[(long)(k0 + g * 16 + khalf + j) * ld + col];
  return r;
}

__device__ __forceinline__ void atomic_absmax(float* p, float v) {
  atomicMax((unsigned int*)p, __float_as_uint(fabsf(v)));
}

__device__ __forceinline__ uint32_t pcg_hash(uint32_t v) {
  v = v * 747796405u + 2891336453u;
  uint32_t w = ((v >> ((v >> 28u) + 4u)) ^ v) * 277803737u;
  return (w >> 22u) ^ w;
}

// ---------------------------------------------------------------------------
// Generic batched WMMA GEMM:  C = scale * (A[M,K] . W[N,K]^T) + bias
// Batch z: offset = (z/nInner)*sOuter + (z%nInner)*sInner per operand.
// ---------------------------------------------------------------------------
__global__ __launch_bounds__(256) void gemm_nt_kernel(
    const float* A, long lda, long sAo, long sAi,
    const float* B, long ldb, long sBo, long sBi,
    float* C, long ldc, long sCo, long sCi,
    const float* bias, float scale, int M, int N, int K, int nInner) {
  const int lane = threadIdx.x & 31;
  const int wave = threadIdx.x >> 5;
  const int tilesN = N >> 4;
  const int tile = blockIdx.x * (blockDim.x >> 5) + wave;
  if (tile >= (M >> 4) * tilesN) return;
  const int tm = tile / tilesN, tn = tile % tilesN;
  const int z = blockIdx.y;
  const int zo = z / nInner, zi = z % nInner;
  const float* Ab = A + zo * sAo + zi * sAi;
  const float* Bb = B + zo * sBo + zi * sBi;
  float* Cb = C + zo * sCo + zi * sCi;
  const int m0 = tm * 16, n0 = tn * 16;
  v8f acc = vzero8();
  for (int k0 = 0; k0 < K; k0 += 32) {
    v16bf a = frag_row_f32(Ab, m0 + (lane & 15), lda, k0, lane);
    v16bf b = frag_row_f32(Bb, n0 + (lane & 15), ldb, k0, lane);
    acc = wmma_bf16(a, b, acc);
  }
  const int n = lane & 15;
  const int moff = (lane >> 4) * 8;
  const float bv = bias ? bias[n0 + n] : 0.0f;
#pragma unroll
  for (int r = 0; r < 8; ++r)
    Cb[(long)(m0 + moff + r) * ldc + n0 + n] = acc[r] * scale + bv;
}

// C = A[M,K] . B[K,N]  (B row-major, not transposed) -- used for attn @ V
__global__ __launch_bounds__(256) void gemm_nn_kernel(
    const float* A, long lda, long sAo, long sAi,
    const float* B, long ldb, long sBo, long sBi,
    float* C, long ldc, long sCo, long sCi,
    int M, int N, int K, int nInner) {
  const int lane = threadIdx.x & 31;
  const int wave = threadIdx.x >> 5;
  const int tilesN = N >> 4;
  const int tile = blockIdx.x * (blockDim.x >> 5) + wave;
  if (tile >= (M >> 4) * tilesN) return;
  const int tm = tile / tilesN, tn = tile % tilesN;
  const int z = blockIdx.y;
  const int zo = z / nInner, zi = z % nInner;
  const float* Ab = A + zo * sAo + zi * sAi;
  const float* Bb = B + zo * sBo + zi * sBi;
  float* Cb = C + zo * sCo + zi * sCi;
  const int m0 = tm * 16, n0 = tn * 16;
  v8f acc = vzero8();
  for (int k0 = 0; k0 < K; k0 += 32) {
    v16bf a = frag_row_f32(Ab, m0 + (lane & 15), lda, k0, lane);
    v16bf b = frag_nn_f32(Bb, k0, ldb, n0 + (lane & 15), lane);
    acc = wmma_bf16(a, b, acc);
  }
  const int n = lane & 15;
  const int moff = (lane >> 4) * 8;
#pragma unroll
  for (int r = 0; r < 8; ++r)
    Cb[(long)(m0 + moff + r) * ldc + n0 + n] = acc[r];
}

// ---------------------------------------------------------------------------
// Softmax over rows of length 256 (one block / row)
// ---------------------------------------------------------------------------
__global__ __launch_bounds__(256) void softmax_kernel(float* s) {
  const long row = blockIdx.x;
  const int tid = threadIdx.x;
  const int lane = tid & 31, wv = tid >> 5;
  float v = s[row * 256 + tid];
  float m = v;
#pragma unroll
  for (int o = 16; o > 0; o >>= 1) m = fmaxf(m, __shfl_down(m, o));
  __shared__ float redm[8];
  __shared__ float reds[8];
  if (lane == 0) redm[wv] = m;
  __syncthreads();
  if (tid == 0) {
    float mm = redm[0];
    for (int i = 1; i < 8; ++i) mm = fmaxf(mm, redm[i]);
    redm[0] = mm;
  }
  __syncthreads();
  m = redm[0];
  float e = __expf(v - m);
  float sum = e;
#pragma unroll
  for (int o = 16; o > 0; o >>= 1) sum += __shfl_down(sum, o);
  if (lane == 0) reds[wv] = sum;
  __syncthreads();
  if (tid == 0) {
    float ss = 0.0f;
    for (int i = 0; i < 8; ++i) ss += reds[i];
    reds[0] = ss;
  }
  __syncthreads();
  s[row * 256 + tid] = e / reds[0];
}

// ---------------------------------------------------------------------------
// out[row] = LayerNorm(x[row] + y[row]) * g + b   (E=768, block=256)
// ---------------------------------------------------------------------------
__global__ __launch_bounds__(256) void add_ln_kernel(const float* x, const float* y,
                                                     const float* g, const float* b,
                                                     float* out) {
  const long row = blockIdx.x;
  const int tid = threadIdx.x;
  const int lane = tid & 31, wv = tid >> 5;
  const float* xr = x + row * EMBED;
  const float* yr = y + row * EMBED;
  float v[3], s = 0.0f, ss = 0.0f;
#pragma unroll
  for (int j = 0; j < 3; ++j) {
    const int e = tid + j * 256;
    float t = xr[e] + yr[e];
    v[j] = t; s += t; ss += t * t;
  }
#pragma unroll
  for (int o = 16; o > 0; o >>= 1) { s += __shfl_down(s, o); ss += __shfl_down(ss, o); }
  __shared__ float sums[8];
  __shared__ float sqs[8];
  if (lane == 0) { sums[wv] = s; sqs[wv] = ss; }
  __syncthreads();
  if (tid == 0) {
    float a = 0.0f, q = 0.0f;
    for (int i = 0; i < 8; ++i) { a += sums[i]; q += sqs[i]; }
    sums[0] = a; sqs[0] = q;
  }
  __syncthreads();
  const float mean = sums[0] * (1.0f / EMBED);
  const float var = sqs[0] * (1.0f / EMBED) - mean * mean;
  const float inv = rsqrtf(var + 1e-5f);
#pragma unroll
  for (int j = 0; j < 3; ++j) {
    const int e = tid + j * 256;
    out[row * EMBED + e] = (v[j] - mean) * inv * g[e] + b[e];
  }
}

// ---------------------------------------------------------------------------
// qlinear preparation kernels
// ---------------------------------------------------------------------------
__global__ void init_scalars_kernel(float* scal) {
  if (threadIdx.x < 16) scal[threadIdx.x] = 0.0f;
}

__global__ __launch_bounds__(256) void absmax_kernel(const float* v, long n, float* outp) {
  long i = (long)blockIdx.x * blockDim.x + threadIdx.x;
  const long stride = (long)gridDim.x * blockDim.x;
  float m = 0.0f;
  for (; i < n; i += stride) m = fmaxf(m, fabsf(v[i]));
  atomic_absmax(outp, m);
}

// conductance planes: cond[k][o*Fin+f] = (bit_k*(1-1/r)+1/r)*(1+0.02*N(0,1))
__global__ __launch_bounds__(256) void build_cond_kernel(const float* W, long n,
                                                         const float* scal, int slot,
                                                         __bf16* cond) {
  float wmax = scal[slot];
  if (!(wmax > 0.0f)) wmax = 1.0f;
  long i = (long)blockIdx.x * blockDim.x + threadIdx.x;
  const long stride = (long)gridDim.x * blockDim.x;
  for (; i < n; i += stride) {
    const float w = W[i];
    int wd = (int)rintf((w / wmax + 1.0f) * 0.5f * 15.0f);
    wd = wd < 0 ? 0 : (wd > 15 ? 15 : wd);
#pragma unroll
    for (int k = 0; k < 4; ++k) {
      const float bit = (float)((wd >> k) & 1);
      const float gcc = bit * 0.9f + 0.1f;
      const uint32_t h1 = pcg_hash((uint32_t)(i * 4 + k) ^ 0x9e3779b9u);
      const uint32_t h2 = pcg_hash(h1 ^ 0x85ebca6bu);
      const float u1 = ((float)h1 + 0.5f) * (1.0f / 4294967296.0f);
      const float u2 = ((float)h2 + 0.5f) * (1.0f / 4294967296.0f);
      const float nrm = sqrtf(-2.0f * __logf(u1)) * __cosf(6.2831853f * u2);
      cond[(long)k * n + i] = (__bf16)(gcc * (1.0f + 0.02f * nrm));
    }
  }
}

// signed activation bit planes: xbits[z][t*Fin+f] = sign(Xq)*bit_z(|Xq|)
__global__ __launch_bounds__(256) void build_xbits_kernel(const float* x, long n,
                                                          const float* scal, int slot,
                                                          __bf16* xbits) {
  float xmax = scal[slot];
  if (!(xmax > 0.0f)) xmax = 1.0f;
  long i = (long)blockIdx.x * blockDim.x + threadIdx.x;
  const long stride = (long)gridDim.x * blockDim.x;
  for (; i < n; i += stride) {
    const float v = x[i];
    const int q = (int)rintf(v / xmax * 7.0f);
    const int a = q < 0 ? -q : q;
    const float sg = (q > 0) ? 1.0f : ((q < 0) ? -1.0f : 0.0f);
#pragma unroll
    for (int z = 0; z < 3; ++z)
      xbits[(long)z * n + i] = (__bf16)((float)((a >> z) & 1) * sg);
  }
}

// dummyP[(t*nc+c)*3+z] = 0.55 * sum_{s<128} xbits[z][t, c*128+s];  also dmax
__global__ __launch_bounds__(256) void dummy_kernel(const __bf16* xbits, int T, int Fin,
                                                    float* dummyP, float* dmaxp) {
  const int nc = Fin >> 7;
  const long total = (long)T * nc * 3;
  long idx = (long)blockIdx.x * blockDim.x + threadIdx.x;
  if (idx >= total) return;
  const int z = (int)(idx % 3);
  const long tc = idx / 3;
  const int c = (int)(tc % nc);
  const long t = tc / nc;
  const __bf16* p = xbits + (long)z * T * Fin + t * Fin + c * 128;
  float s = 0.0f;
  for (int i = 0; i < 128; ++i) s += (float)p[i];
  const float d = 0.55f * s;
  dummyP[idx] = d;
  atomic_absmax(dmaxp, d);
}

// ---------------------------------------------------------------------------
// qlinear pass 1: chunked bit-sliced partial sums via WMMA; the 16x128x3
// xbits tile shared by the block's 8 waves is staged into LDS with
// global_load_async_to_lds_b128 (ASYNCcnt). cond fragments are loaded once
// per (c,k) and reused across the 3 activation-bit planes.
// Grid must cover tiles exactly (barriers are convergent).
// ---------------------------------------------------------------------------
__global__ __launch_bounds__(256) void qpass1_kernel(const __bf16* xbits,
                                                     const __bf16* cond,
                                                     int T, int Fin, int Fout,
                                                     float* ymaxp) {
  const int tid = threadIdx.x;
  const int lane = tid & 31;
  const int wave = tid >> 5;
  const int tilesN = Fout >> 4;           // multiple of 8
  const int tile = blockIdx.x * 8 + wave;
  const int o0 = (tile % tilesN) * 16;
  const int t0blk = ((blockIdx.x * 8) / tilesN) * 16;   // same for all 8 waves
  const int nc = Fin >> 7;
  const long planeX = (long)T * Fin;
  const long planeC = (long)Fout * Fin;

  __shared__ __bf16 sA[3 * 16 * 128];     // [z][row][col] 12 KB
  const int srow = tid >> 4;              // 0..15
  const int scol = (tid & 15) << 3;       // 0..120 (16B chunks)

  float mx = 0.0f;
  for (int c = 0; c < nc; ++c) {
    const int f0 = c << 7;
    __syncthreads();                      // previous chunk fully consumed
#pragma unroll
    for (int z = 0; z < 3; ++z) {
      const __bf16* gp = xbits + (long)z * planeX +
                         (long)(t0blk + srow) * Fin + f0 + scol;
      lds_bf16* lp = (lds_bf16*)(sA + z * 2048 + srow * 128 + scol);
      async_g2l_b128(lp, gp);
    }
    wait_asynccnt0();
    __syncthreads();

    v16bf a[3][4];
#pragma unroll
    for (int z = 0; z < 3; ++z)
#pragma unroll
      for (int sgm = 0; sgm < 4; ++sgm)
        a[z][sgm] = frag_row_bf16(sA + z * 2048, lane & 15, 128, 32 * sgm, lane);

#pragma unroll
    for (int k = 0; k < 4; ++k) {
      const __bf16* Bk = cond + (long)k * planeC;
      v16bf b[4];
#pragma unroll
      for (int sgm = 0; sgm < 4; ++sgm)
        b[sgm] = frag_row_bf16(Bk, o0 + (lane & 15), Fin, f0 + 32 * sgm, lane);
#pragma unroll
      for (int z = 0; z < 3; ++z) {
        v8f p = vzero8();
#pragma unroll
        for (int sgm = 0; sgm < 4; ++sgm) p = wmma_bf16(a[z][sgm], b[sgm], p);
#pragma unroll
        for (int r = 0; r < 8; ++r) mx = fmaxf(mx, fabsf(p[r]));
      }
    }
  }
  atomic_absmax(ymaxp, mx);
}

// ---------------------------------------------------------------------------
// qlinear pass 2: recompute partials (same staging), ADC-quantize, recenter
// with quantized dummy column, shift-add over (z,k), scale, bias, opt. GELU.
// scal slots: [base+0]=wmax [base+1]=xmax [base+2]=ymax [base+3]=dmax
// ---------------------------------------------------------------------------
__global__ __launch_bounds__(256) void qpass2_kernel(const __bf16* xbits,
                                                     const __bf16* cond,
                                                     const float* dummyP,
                                                     const float* scal, int sbase,
                                                     const float* bias, float* out,
                                                     int T, int Fin, int Fout,
                                                     int doGelu) {
  const int tid = threadIdx.x;
  const int lane = tid & 31;
  const int wave = tid >> 5;
  const int tilesN = Fout >> 4;
  const int tile = blockIdx.x * 8 + wave;
  const int o0 = (tile % tilesN) * 16;
  const int t0blk = ((blockIdx.x * 8) / tilesN) * 16;
  const int nc = Fin >> 7;
  const long planeX = (long)T * Fin;
  const long planeC = (long)Fout * Fin;

  const float wmax = scal[sbase + 0];
  const float xmax = scal[sbase + 1];
  float step = scal[sbase + 2] * (1.0f / 31.0f);
  if (!(step > 0.0f)) step = 1.0f;
  float dstep = scal[sbase + 3] * (1.0f / 31.0f);
  if (!(dstep > 0.0f)) dstep = 1.0f;
  // out = acc * (xmax/Xlev) * (2*wmax / ((upper-lower)*Wlev))
  const float alpha = (xmax * (1.0f / 7.0f)) * (2.0f * wmax / (0.9f * 15.0f));

  __shared__ __bf16 sA[3 * 16 * 128];
  const int srow = tid >> 4;
  const int scol = (tid & 15) << 3;
  const int moff = (lane >> 4) * 8;

  float acc[8];
#pragma unroll
  for (int r = 0; r < 8; ++r) acc[r] = 0.0f;

  for (int c = 0; c < nc; ++c) {
    const int f0 = c << 7;
    __syncthreads();
#pragma unroll
    for (int z = 0; z < 3; ++z) {
      const __bf16* gp = xbits + (long)z * planeX +
                         (long)(t0blk + srow) * Fin + f0 + scol;
      lds_bf16* lp = (lds_bf16*)(sA + z * 2048 + srow * 128 + scol);
      async_g2l_b128(lp, gp);
    }
    wait_asynccnt0();
    __syncthreads();

    v16bf a[3][4];
#pragma unroll
    for (int z = 0; z < 3; ++z)
#pragma unroll
      for (int sgm = 0; sgm < 4; ++sgm)
        a[z][sgm] = frag_row_bf16(sA + z * 2048, lane & 15, 128, 32 * sgm, lane);

    float dq[3][8];
#pragma unroll
    for (int z = 0; z < 3; ++z)
#pragma unroll
      for (int r = 0; r < 8; ++r) {
        const float dp = dummyP[((long)(t0blk + moff + r) * nc + c) * 3 + z];
        dq[z][r] = rintf(dp / dstep) * dstep;
      }

#pragma unroll
    for (int k = 0; k < 4; ++k) {
      const __bf16* Bk = cond + (long)k * planeC;
      v16bf b[4];
#pragma unroll
      for (int sgm = 0; sgm < 4; ++sgm)
        b[sgm] = frag_row_bf16(Bk, o0 + (lane & 15), Fin, f0 + 32 * sgm, lane);
#pragma unroll
      for (int z = 0; z < 3; ++z) {
        v8f p = vzero8();
#pragma unroll
        for (int sgm = 0; sgm < 4; ++sgm) p = wmma_bf16(a[z][sgm], b[sgm], p);
        const float sc = (float)(1 << (z + k));
#pragma unroll
        for (int r = 0; r < 8; ++r)
          acc[r] += (rintf(p[r] / step) * step - dq[z][r]) * sc;
      }
    }
  }
  const int n = lane & 15;
  const float bv = bias[o0 + n];
#pragma unroll
  for (int r = 0; r < 8; ++r) {
    float v = acc[r] * alpha + bv;
    if (doGelu) v = 0.5f * v * (1.0f + erff(v * 0.70710678118654752f));
    out[(long)(t0blk + moff + r) * Fout + o0 + n] = v;
  }
}

// ---------------------------------------------------------------------------
// Host-side orchestration
// ---------------------------------------------------------------------------
extern "C" void kernel_launch(void* const* d_in, const int* in_sizes, int n_in,
                              void* d_out, int out_size, void* d_ws, size_t ws_size,
                              hipStream_t stream) {
  (void)in_sizes; (void)n_in; (void)out_size; (void)ws_size;
  const float* x    = (const float*)d_in[0];
  const float* Wqkv = (const float*)d_in[1];
  const float* bqkv = (const float*)d_in[2];
  const float* Wout = (const float*)d_in[3];
  const float* bout = (const float*)d_in[4];
  const float* ln1g = (const float*)d_in[5];
  const float* ln1b = (const float*)d_in[6];
  const float* W1   = (const float*)d_in[7];
  const float* b1   = (const float*)d_in[8];
  const float* W2   = (const float*)d_in[9];
  const float* b2   = (const float*)d_in[10];
  const float* ln2g = (const float*)d_in[11];
  const float* ln2b = (const float*)d_in[12];
  float* out = (float*)d_out;

  // ---- workspace carve-up (~74 MB, all 256B aligned) ----
  char* w = (char*)d_ws;
  float* scal   = (float*)w;                      // 16 scalar slots
  float* qkv    = (float*)(w + 256);              // [512,2304]
  float* scores = qkv + (long)TOK * QKVD;         // [24,256,256]
  float* ctx    = scores + 24L * SEQ * SEQ;       // [512,768]
  float* abuf   = ctx + (long)TOK * EMBED;        // [512,768]
  float* hbuf   = abuf + (long)TOK * EMBED;       // [512,768]
  float* m1     = hbuf + (long)TOK * EMBED;       // [512,3072]
  float* m2     = m1 + (long)TOK * MLPD;          // [512,768]
  float* dummy1 = m2 + (long)TOK * EMBED;         // [512,6,3]
  float* dummy2 = dummy1 + (long)TOK * 6 * 3;     // [512,24,3]
  __bf16* xb1   = (__bf16*)(dummy2 + (long)TOK * 24 * 3); // [3][512,768]
  __bf16* xb2   = xb1 + 3L * TOK * EMBED;                 // [3][512,3072]
  __bf16* cond1 = xb2 + 3L * TOK * MLPD;                  // [4][3072,768]
  __bf16* cond2 = cond1 + 4L * MLPD * EMBED;              // [4][768,3072]

  init_scalars_kernel<<<1, 32, 0, stream>>>(scal);

  // ---- attention ----
  // QKV = X[512,768] . Wqkv[2304,768]^T + b : 4608 tiles
  gemm_nt_kernel<<<dim3(576, 1), 256, 0, stream>>>(
      x, EMBED, 0, 0, Wqkv, EMBED, 0, 0, qkv, QKVD, 0, 0,
      bqkv, 1.0f, TOK, QKVD, EMBED, 1);
  // scores[n,h,l,m] = (q . k^T)/8 : 24 batches of 256x256x64
  gemm_nt_kernel<<<dim3(32, 24), 256, 0, stream>>>(
      qkv, LDQKV, QKVD, 64,
      qkv + EMBED, LDQKV, QKVD, 64,
      scores, SEQ, 12L * SEQ * SEQ, (long)SEQ * SEQ,
      nullptr, 0.125f, SEQ, SEQ, 64, HEADS);
  softmax_kernel<<<24 * SEQ, 256, 0, stream>>>(scores);
  // ctx[l,n,h,d] = attn . V : 24 batches of 256x64x256
  gemm_nn_kernel<<<dim3(8, 24), 256, 0, stream>>>(
      scores, SEQ, 12L * SEQ * SEQ, (long)SEQ * SEQ,
      qkv + 2 * EMBED, LDQKV, QKVD, 64,
      ctx, (long)BATCH * EMBED, EMBED, 64,
      SEQ, 64, SEQ, HEADS);
  // a = ctx . Wout^T + b
  gemm_nt_kernel<<<dim3(192, 1), 256, 0, stream>>>(
      ctx, EMBED, 0, 0, Wout, EMBED, 0, 0, abuf, EMBED, 0, 0,
      bout, 1.0f, TOK, EMBED, EMBED, 1);
  // h = LN1(x + a)
  add_ln_kernel<<<TOK, 256, 0, stream>>>(x, abuf, ln1g, ln1b, hbuf);

  // ---- qlinear 1: h[512,768] -> m1[512,3072], fused exact GELU ----
  absmax_kernel<<<1024, 256, 0, stream>>>(W1, (long)MLPD * EMBED, scal + 0);
  absmax_kernel<<<768, 256, 0, stream>>>(hbuf, (long)TOK * EMBED, scal + 1);
  build_cond_kernel<<<4096, 256, 0, stream>>>(W1, (long)MLPD * EMBED, scal, 0, cond1);
  build_xbits_kernel<<<1536, 256, 0, stream>>>(hbuf, (long)TOK * EMBED, scal, 1, xb1);
  dummy_kernel<<<36, 256, 0, stream>>>(xb1, TOK, EMBED, dummy1, scal + 3);
  // 6144 tiles exactly = 768 blocks * 8 waves
  qpass1_kernel<<<768, 256, 0, stream>>>(xb1, cond1, TOK, EMBED, MLPD, scal + 2);
  qpass2_kernel<<<768, 256, 0, stream>>>(xb1, cond1, dummy1, scal, 0, b1, m1,
                                         TOK, EMBED, MLPD, 1);

  // ---- qlinear 2: m1[512,3072] -> m2[512,768] ----
  absmax_kernel<<<1024, 256, 0, stream>>>(W2, (long)EMBED * MLPD, scal + 4);
  absmax_kernel<<<1024, 256, 0, stream>>>(m1, (long)TOK * MLPD, scal + 5);
  build_cond_kernel<<<4096, 256, 0, stream>>>(W2, (long)EMBED * MLPD, scal, 4, cond2);
  build_xbits_kernel<<<6144, 256, 0, stream>>>(m1, (long)TOK * MLPD, scal, 5, xb2);
  dummy_kernel<<<144, 256, 0, stream>>>(xb2, TOK, MLPD, dummy2, scal + 7);
  // 1536 tiles exactly = 192 blocks * 8 waves
  qpass1_kernel<<<192, 256, 0, stream>>>(xb2, cond2, TOK, MLPD, EMBED, scal + 6);
  qpass2_kernel<<<192, 256, 0, stream>>>(xb2, cond2, dummy2, scal, 4, b2, m2,
                                         TOK, MLPD, EMBED, 0);

  // ---- final: out = LN2(h + m2) ----
  add_ln_kernel<<<TOK, 256, 0, stream>>>(hbuf, m2, ln2g, ln2b, out);
}
